// Network_74320114090279
// MI455X (gfx1250) — matrix operand order repaired and checked by reading, not compiled, not run on hardware
//
#include <hip/hip_runtime.h>

// ---------------------------------------------------------------------------
// AXform point-cloud network for gfx1250 (MI455X).
// All einsum/matmul work runs through one batched WMMA GEMM kernel using
// v_wmma_f32_16x16x32_bf16 (f32 accumulate, native f32->bf16 converts),
// N-blocked 4 tiles/wave with float4 A-fragment loads.
// FPS runs one workgroup per point set with the running-distance array in LDS.
// BN stats use a two-stage deterministic tree reduction.
// ---------------------------------------------------------------------------

#define G    16
#define BSZ  16
#define NPS  512      // NP
#define EPSF 1e-5f

typedef __attribute__((ext_vector_type(16))) __bf16 v16bf;
typedef __attribute__((ext_vector_type(8)))  float  v8f;

// ---------------------------------------------------------------------------
// Batched GEMM: C[s] = A[s] (MxK) @ B[s] (KxN) + bias[s]
// One wave per 16 x (16*NB) output strip; K-loop in steps of 32 via WMMA.
// TRANSA: A element (m,k) read at A[k*lda + m] (used for w^T @ xb).
// ---------------------------------------------------------------------------
template<int TRANSA, int NB>
__global__ void wmma_gemm(const float* __restrict__ A, const float* __restrict__ Bm,
                          const float* __restrict__ bias, float* __restrict__ C,
                          int M, int N, int K, int lda, int ldb, int ldc,
                          long long sA, long long sB, long long sBias, long long sC)
{
    int z = blockIdx.z;
    A  += (long long)z * sA;
    Bm += (long long)z * sB;
    C  += (long long)z * sC;
    if (bias) bias += (long long)z * sBias;

    int tileM = blockIdx.x * 16;
    int tileN = blockIdx.y * (16 * NB);
    int lane  = threadIdx.x & 31;
    int l     = lane & 15;
    bool hi   = lane >= 16;
    int row   = tileM + l;     // A rows: both lane halves cover M=0..15 (diff K)

    v8f acc[NB];
#pragma unroll
    for (int nb = 0; nb < NB; ++nb) acc[nb] = (v8f){0.f,0.f,0.f,0.f,0.f,0.f,0.f,0.f};

    for (int k0 = 0; k0 < K; k0 += 32) {
        // ---- A fragment (ISA 7.12.2, 16-bit A 16x32):
        //  lanes 0-15 : halves -> K = k0 + {0..7, 16..23}
        //  lanes 16-31: halves -> K = k0 + {8..15, 24..31}
        v16bf a;
        if (TRANSA) {
#pragma unroll
            for (int i = 0; i < 16; ++i) {
                int kk = (i < 8) ? i : i + 8;
                if (hi) kk += 8;
                a[i] = (__bf16)A[(long long)(k0 + kk) * lda + row];
            }
        } else {
            const float* Ap = A + (long long)row * lda + k0 + (hi ? 8 : 0);
            float4 f0 = *(const float4*)(Ap);
            float4 f1 = *(const float4*)(Ap + 4);
            float4 f2 = *(const float4*)(Ap + 16);
            float4 f3 = *(const float4*)(Ap + 20);
            a[0] = (__bf16)f0.x; a[1] = (__bf16)f0.y; a[2]  = (__bf16)f0.z; a[3]  = (__bf16)f0.w;
            a[4] = (__bf16)f1.x; a[5] = (__bf16)f1.y; a[6]  = (__bf16)f1.z; a[7]  = (__bf16)f1.w;
            a[8] = (__bf16)f2.x; a[9] = (__bf16)f2.y; a[10] = (__bf16)f2.z; a[11] = (__bf16)f2.w;
            a[12]= (__bf16)f3.x; a[13]= (__bf16)f3.y; a[14] = (__bf16)f3.z; a[15] = (__bf16)f3.w;
        }
        // ---- B fragments (KxN): lanes 0-15 hold K=k0+{0..7,16..23}, lanes 16-31 K=k0+{8..15,24..31}
#pragma unroll
        for (int nb = 0; nb < NB; ++nb) {
            int col = tileN + nb * 16 + l;
            const float* Bp = Bm + (long long)(k0 + (hi ? 8 : 0)) * ldb + col;
            v16bf b;
#pragma unroll
            for (int i = 0; i < 16; ++i)
                b[i] = (__bf16)Bp[(long long)((i < 8) ? i : i + 8) * ldb];
            acc[nb] = __builtin_amdgcn_wmma_f32_16x16x32_bf16(
                          false, a, false, b, (short)0, acc[nb], false, false);
        }
    }

#pragma unroll
    for (int nb = 0; nb < NB; ++nb) {
        int col = tileN + nb * 16 + l;
        float bv = bias ? bias[col] : 0.f;
#pragma unroll
        for (int r = 0; r < 8; ++r) {
            int m = tileM + (hi ? 8 : 0) + r;   // C layout: VGPR r -> M=r / M=8+r
            C[(long long)m * ldc + col] = acc[nb][r] + bv;
        }
    }
}

// ---------------------------------------------------------------------------
// Farthest point sampling: one 256-thread workgroup per point set.
// Running min-distance array lives in LDS; argmax via LDS tree reduction
// (ties -> lowest index, matching jnp.argmax).
// ---------------------------------------------------------------------------
__global__ void fps_kernel(const float* __restrict__ pts, int P, int inStride,
                           int npoint, float* __restrict__ out, int outStride)
{
    __shared__ float dist[2048];
    __shared__ float rv[256];
    __shared__ int   ri[256];
    const float* p = pts + (long long)blockIdx.x * inStride;
    float* o = out + (long long)blockIdx.x * outStride;
    int tid = threadIdx.x;

    for (int i = tid; i < P; i += blockDim.x) dist[i] = 1e10f;
    if (tid == 0) { o[0] = p[0]; o[1] = p[1]; o[2] = p[2]; }
    int cur = 0;
    __syncthreads();

    for (int it = 1; it < npoint; ++it) {
        float cx = p[cur * 3], cy = p[cur * 3 + 1], cz = p[cur * 3 + 2];
        float best = -1.f; int bi = 0;
        for (int j = tid; j < P; j += blockDim.x) {
            float dx = p[j * 3] - cx, dy = p[j * 3 + 1] - cy, dz = p[j * 3 + 2] - cz;
            float d  = dx * dx + dy * dy + dz * dz;
            float nd = fminf(dist[j], d);
            dist[j] = nd;
            if (nd > best) { best = nd; bi = j; }
        }
        rv[tid] = best; ri[tid] = bi;
        __syncthreads();
        for (int s = 128; s > 0; s >>= 1) {
            if (tid < s) {
                if (rv[tid + s] > rv[tid] ||
                    (rv[tid + s] == rv[tid] && ri[tid + s] < ri[tid])) {
                    rv[tid] = rv[tid + s]; ri[tid] = ri[tid + s];
                }
            }
            __syncthreads();
        }
        cur = ri[0];
        if (tid == 0) {
            o[it * 3] = p[cur * 3]; o[it * 3 + 1] = p[cur * 3 + 1]; o[it * 3 + 2] = p[cur * 3 + 2];
        }
        __syncthreads();
    }
}

// pointnet layer 1: (rows,3) @ (3,64) + b  (K=3 too small for WMMA)
__global__ void linear3_kernel(const float* __restrict__ x, const float* __restrict__ w,
                               const float* __restrict__ b, float* __restrict__ z,
                               long long rows)
{
    long long idx = (long long)blockIdx.x * blockDim.x + threadIdx.x;
    if (idx >= rows * 64) return;
    int c = (int)(idx & 63);
    long long r = idx >> 6;
    float acc = b[c];
    acc += x[r * 3 + 0] * w[0 * 64 + c];
    acc += x[r * 3 + 1] * w[1 * 64 + c];
    acc += x[r * 3 + 2] * w[2 * 64 + c];
    z[idx] = acc;
}

// ---- BatchNorm stats: deterministic two-stage reduction --------------------
// Stage 1: each block handles one (set, row-chunk); per-channel partial
// sum/sumsq written to a fixed slot (no atomics -> bitwise deterministic).
__global__ void bn_stats_part(const float* __restrict__ X, int M, int C, int nchunk,
                              float* __restrict__ part)
{
    int s  = blockIdx.x / nchunk;
    int ch = blockIdx.x % nchunk;
    int rows = (M + nchunk - 1) / nchunk;
    int r0 = ch * rows;
    int r1 = min(M, r0 + rows);
    const float* Xs = X + (long long)s * M * C;
    float* ps = part + (long long)blockIdx.x * C * 2;
    for (int c = threadIdx.x; c < C; c += blockDim.x) {
        float sum = 0.f, sq = 0.f;
        for (int m = r0; m < r1; ++m) {
            float v = Xs[(long long)m * C + c];
            sum += v; sq += v * v;
        }
        ps[c * 2]     = sum;
        ps[c * 2 + 1] = sq;
    }
}

// Stage 2: fixed-order sum across chunks -> mean / inv-std (biased var).
__global__ void bn_stats_final(const float* __restrict__ part, int C, int nchunk, int M,
                               float* __restrict__ stats)
{
    int s = blockIdx.x;
    for (int c = threadIdx.x; c < C; c += blockDim.x) {
        float sum = 0.f, sq = 0.f;
        for (int k = 0; k < nchunk; ++k) {
            const float* ps = part + (long long)(s * nchunk + k) * C * 2;
            sum += ps[c * 2];
            sq  += ps[c * 2 + 1];
        }
        float mean = sum / (float)M;
        float var  = sq / (float)M - mean * mean;
        stats[((long long)s * C + c) * 2]     = mean;
        stats[((long long)s * C + c) * 2 + 1] = rsqrtf(var + EPSF);
    }
}

// In-place normalize (+ optional ReLU)
__global__ void bn_act_kernel(float* __restrict__ X, const float* __restrict__ stats,
                              long long total, int M, int C, int relu)
{
    long long idx = (long long)blockIdx.x * blockDim.x + threadIdx.x;
    if (idx >= total) return;
    int c = (int)(idx % C);
    long long s = idx / ((long long)M * C);
    float mean = stats[(s * C + c) * 2];
    float rstd = stats[(s * C + c) * 2 + 1];
    float v = (X[idx] - mean) * rstd;
    if (relu) v = fmaxf(v, 0.f);
    X[idx] = v;
}

// pointnet tail: normalize then max over P points -> feat (B,C)
__global__ void bn_maxpool_kernel(const float* __restrict__ Z, const float* __restrict__ stats,
                                  int Pn, int C, float* __restrict__ feat)
{
    int b = blockIdx.x;
    const float* zb = Z + (long long)b * Pn * C;
    for (int c = threadIdx.x; c < C; c += blockDim.x) {
        float mean = stats[c * 2], rstd = stats[c * 2 + 1];
        float mx = -3.4e38f;
        for (int p = 0; p < Pn; ++p)
            mx = fmaxf(mx, (zb[(long long)p * C + c] - mean) * rstd);
        feat[(long long)b * C + c] = mx;
    }
}

// In-place softmax over dim n of H(S, 128, 1024); one block per set s,
// threads stride over the 1024 e-columns (coalesced).
__global__ void softmax_kernel(float* __restrict__ H)
{
    int s = blockIdx.x;
    float* h = H + (long long)s * 128 * 1024;
    int t = threadIdx.x;                    // 256 threads, 4 columns each
    float mx[4] = {-3.4e38f, -3.4e38f, -3.4e38f, -3.4e38f};
    for (int n = 0; n < 128; ++n)
#pragma unroll
        for (int j = 0; j < 4; ++j)
            mx[j] = fmaxf(mx[j], h[n * 1024 + t + j * 256]);
    float sm[4] = {0.f, 0.f, 0.f, 0.f};
    for (int n = 0; n < 128; ++n)
#pragma unroll
        for (int j = 0; j < 4; ++j)
            sm[j] += expf(h[n * 1024 + t + j * 256] - mx[j]);
    for (int n = 0; n < 128; ++n)
#pragma unroll
        for (int j = 0; j < 4; ++j) {
            int o = n * 1024 + t + j * 256;
            h[o] = expf(h[o] - mx[j]) / sm[j];
        }
}

// final 1x1 conv (K2=32 -> 3) + optional residual addend (x_base)
__global__ void c4_kernel(const float* __restrict__ y, const float* __restrict__ w,
                          const float* __restrict__ bias, const float* __restrict__ addend,
                          float* __restrict__ out, long long total)
{
    long long idx = (long long)blockIdx.x * blockDim.x + threadIdx.x;
    if (idx >= total) return;                       // total = G*B*1024
    int g = (int)(idx / (BSZ * 1024));
    const float* yr = y + idx * 32;
    float o0 = bias[g * 3 + 0], o1 = bias[g * 3 + 1], o2 = bias[g * 3 + 2];
#pragma unroll
    for (int k = 0; k < 32; ++k) {
        float v = yr[k];
        o0 += v * w[g * 96 + k * 3 + 0];
        o1 += v * w[g * 96 + k * 3 + 1];
        o2 += v * w[g * 96 + k * 3 + 2];
    }
    if (addend) {
        o0 += addend[idx * 3 + 0]; o1 += addend[idx * 3 + 1]; o2 += addend[idx * 3 + 2];
    }
    out[idx * 3 + 0] = o0; out[idx * 3 + 1] = o1; out[idx * 3 + 2] = o2;
}

// (G,B,1024,3) -> (B, G*1024, 3)
__global__ void gather_out_kernel(const float* __restrict__ xg, float* __restrict__ out,
                                  long long total)
{
    long long idx = (long long)blockIdx.x * blockDim.x + threadIdx.x;
    if (idx >= total) return;                       // total = G*B*1024*3
    int j = (int)(idx % 3);
    long long e3 = idx / 3;
    int e = (int)(e3 % 1024);
    long long gb = e3 / 1024;
    int b = (int)(gb % BSZ);
    int g = (int)(gb / BSZ);
    out[(((long long)b * G + g) * 1024 + e) * 3 + j] = xg[idx];
}

// x_base second half: broadcast x_partial(B,512,3) into (G,B,[512:1024],3)
__global__ void fill_base_kernel(const float* __restrict__ xpart, float* __restrict__ xbase,
                                 long long total)
{
    long long idx = (long long)blockIdx.x * blockDim.x + threadIdx.x;
    if (idx >= total) return;                       // total = G*B*512*3
    int j = (int)(idx % 3);
    long long e3 = idx / 3;
    int e = (int)(e3 % 512);
    long long gb = e3 / 512;
    int b = (int)(gb % BSZ);
    int g = (int)(gb / BSZ);
    xbase[(((long long)g * BSZ + b) * 1024 + 512 + e) * 3 + j] =
        xpart[((long long)b * 512 + e) * 3 + j];
}

// ---------------------------------------------------------------------------
// Host-side helpers
// ---------------------------------------------------------------------------
static void gemm(hipStream_t st, const float* A, const float* B, const float* bias, float* C,
                 int M, int N, int K, int lda, int ldb, int ldc,
                 long long sA, long long sB, long long sBias, long long sC,
                 int batch, int transA)
{
    if (transA) {   // only the attention contraction: N multiple of 32
        dim3 grid(M / 16, N / 32, batch);
        wmma_gemm<1, 2><<<grid, 32, 0, st>>>(A, B, bias, C, M, N, K, lda, ldb, ldc,
                                             sA, sB, sBias, sC);
    } else {        // all other GEMMs: N multiple of 64
        dim3 grid(M / 16, N / 64, batch);
        wmma_gemm<0, 4><<<grid, 32, 0, st>>>(A, B, bias, C, M, N, K, lda, ldb, ldc,
                                             sA, sB, sBias, sC);
    }
}

static int pick_nchunk(int M) { return (M >= 16384) ? 64 : (M >= 1024) ? 8 : 1; }

static void bn(hipStream_t st, float* X, int S, int M, int C, int relu,
               float* stats, float* part)
{
    int nchunk = pick_nchunk(M);
    bn_stats_part<<<S * nchunk, 256, 0, st>>>(X, M, C, nchunk, part);
    bn_stats_final<<<S, 256, 0, st>>>(part, C, nchunk, M, stats);
    long long t = (long long)S * M * C;
    bn_act_kernel<<<(unsigned)((t + 255) / 256), 256, 0, st>>>(X, stats, t, M, C, relu);
}

struct AxW { const float *fw, *fb, *c1w, *c1b, *c2w, *c2b, *c3w, *c3b, *c4w, *c4b; };

static void axform(hipStream_t st, const float* mfeat, const AxW& W,
                   float* xb, float* ah1, float* ah2, float* h3, float* ybuf,
                   float* stats, float* part, const float* addend, float* outg)
{
    // xb = mfeat @ fw + fb : (G) x (16x128)@(128x4096)
    gemm(st, mfeat, W.fw, W.fb, xb, 16, 4096, 128, 128, 4096, 4096,
         16 * 128, 128 * 4096, 4096, 16 * 4096, G, 0);
    // c1: (G) x (2048x32)@(32x64)
    gemm(st, xb, W.c1w, W.c1b, ah1, 2048, 64, 32, 32, 64, 64,
         2048 * 32, 32 * 64, 64, 2048 * 64, G, 0);
    bn(st, ah1, G, 2048, 64, 1, stats, part);
    // c2: (G) x (2048x64)@(64x128)
    gemm(st, ah1, W.c2w, W.c2b, ah2, 2048, 128, 64, 64, 128, 128,
         2048 * 64, 64 * 128, 128, 2048 * 128, G, 0);
    bn(st, ah2, G, 2048, 128, 1, stats, part);
    // c3: (G) x (2048x128)@(128x1024) -> h3 (no BN), then softmax over n in place
    gemm(st, ah2, W.c3w, W.c3b, h3, 2048, 1024, 128, 128, 1024, 1024,
         2048 * 128, 128 * 1024, 1024, 2048 * 1024, G, 0);
    softmax_kernel<<<G * BSZ, 256, 0, st>>>(h3);
    // y[e,k] = sum_n w[n,e]*xb[n,k] : (G*B) x (1024x128)@(128x32), A transposed
    gemm(st, h3, xb, nullptr, ybuf, 1024, 32, 128, 1024, 32, 32,
         128 * 1024, 4096, 0, 1024 * 32, G * BSZ, 1);
    // c4: (32 -> 3) + optional residual
    long long t = (long long)G * BSZ * 1024;
    c4_kernel<<<(unsigned)((t + 255) / 256), 256, 0, st>>>(ybuf, W.c4w, W.c4b, addend, outg, t);
}

extern "C" void kernel_launch(void* const* d_in, const int* in_sizes, int n_in,
                              void* d_out, int out_size, void* d_ws, size_t ws_size,
                              hipStream_t stream)
{
    (void)in_sizes; (void)n_in; (void)out_size; (void)ws_size;
    const float* x      = (const float*)d_in[0];
    const float* pn_w1  = (const float*)d_in[1];
    const float* pn_b1  = (const float*)d_in[2];
    const float* pn_w2  = (const float*)d_in[3];
    const float* pn_b2  = (const float*)d_in[4];
    const float* pn_w3  = (const float*)d_in[5];
    const float* pn_b3  = (const float*)d_in[6];
    const float* mp_w1  = (const float*)d_in[7];
    const float* mp_b1  = (const float*)d_in[8];
    const float* mp_w2  = (const float*)d_in[9];
    const float* mp_b2  = (const float*)d_in[10];
    const float* mp_w3  = (const float*)d_in[11];
    const float* mp_b3  = (const float*)d_in[12];
    const float* mp_w4  = (const float*)d_in[13];
    const float* mp_b4  = (const float*)d_in[14];
    AxW gw = { (const float*)d_in[15], (const float*)d_in[16], (const float*)d_in[17],
               (const float*)d_in[18], (const float*)d_in[19], (const float*)d_in[20],
               (const float*)d_in[21], (const float*)d_in[22], (const float*)d_in[23],
               (const float*)d_in[24] };
    AxW rw = { (const float*)d_in[25], (const float*)d_in[26], (const float*)d_in[27],
               (const float*)d_in[28], (const float*)d_in[29], (const float*)d_in[30],
               (const float*)d_in[31], (const float*)d_in[32], (const float*)d_in[33],
               (const float*)d_in[34] };
    float* out = (float*)d_out;

    // workspace arena (peak ~236 MB); big buffer shared by pointnet-z3 / axform-h3
    char* base = (char*)d_ws;
    size_t off = 0;
    auto carve = [&](size_t elems) -> float* {
        float* p = (float*)(base + off);
        off += ((elems * sizeof(float) + 255) & ~(size_t)255);
        return p;
    };
    float* bigbuf = carve(32768ull * 1024);          // z3 then h3(gen) then h3(ref)
    float* z1     = carve(32768ull * 64);
    float* z2     = carve(32768ull * 128);
    float* stats  = carve(16ull * 1024 * 2);
    float* part   = carve(128ull * 1024 * 2);        // bn partials (max S*nchunk=128, C=1024)
    float* feat   = carve(16ull * 1024);
    float* mh1    = carve((size_t)G * 16 * 1024);
    float* mh2    = carve((size_t)G * 16 * 1024);
    float* mfeat  = carve((size_t)G * 16 * 128);
    float* xb     = carve((size_t)G * 16 * 4096);
    float* ah1    = carve((size_t)G * 2048 * 64);
    float* ah2    = carve((size_t)G * 2048 * 128);
    float* ybuf   = carve((size_t)G * 16 * 1024 * 32);
    float* x1g    = carve((size_t)G * 16 * 1024 * 3);
    float* x2g    = carve((size_t)G * 16 * 1024 * 3);
    float* xbase  = carve((size_t)G * 16 * 1024 * 3);
    float* xpart  = carve((size_t)16 * 512 * 3);

    const long long ROWS = 16LL * 2048;              // B*P
    const long long T3   = (long long)G * BSZ * 1024 * 3;

    // ---- FPS of input: x_partial (B,512,3)
    fps_kernel<<<16, 256, 0, stream>>>(x, 2048, 2048 * 3, NPS, xpart, NPS * 3);

    // ---- PointNet encoder
    linear3_kernel<<<(unsigned)((ROWS * 64 + 255) / 256), 256, 0, stream>>>(x, pn_w1, pn_b1, z1, ROWS);
    bn(stream, z1, 1, (int)ROWS, 64, 1, stats, part);

    gemm(stream, z1, pn_w2, pn_b2, z2, (int)ROWS, 128, 64, 64, 128, 128, 0, 0, 0, 0, 1, 0);
    bn(stream, z2, 1, (int)ROWS, 128, 1, stats, part);

    gemm(stream, z2, pn_w3, pn_b3, bigbuf, (int)ROWS, 1024, 128, 128, 1024, 1024, 0, 0, 0, 0, 1, 0);
    {   // stats only (no in-place act): normalize happens inside maxpool
        int nchunk = pick_nchunk((int)ROWS);
        bn_stats_part<<<nchunk, 256, 0, stream>>>(bigbuf, (int)ROWS, 1024, nchunk, part);
        bn_stats_final<<<1, 256, 0, stream>>>(part, 1024, nchunk, (int)ROWS, stats);
    }
    bn_maxpool_kernel<<<16, 256, 0, stream>>>(bigbuf, stats, 2048, 1024, feat);   // feat (B,1024)

    // ---- Mapping MLPs (16 branches; A shared via strideA=0 for layer 1)
    gemm(stream, feat, mp_w1, mp_b1, mh1, 16, 1024, 1024, 1024, 1024, 1024,
         0, 1024 * 1024, 1024, 16 * 1024, G, 0);
    bn(stream, mh1, G, 16, 1024, 1, stats, part);

    gemm(stream, mh1, mp_w2, mp_b2, mh2, 16, 1024, 1024, 1024, 1024, 1024,
         16 * 1024, 1024 * 1024, 1024, 16 * 1024, G, 0);
    bn(stream, mh2, G, 16, 1024, 1, stats, part);

    gemm(stream, mh2, mp_w3, mp_b3, mh1, 16, 1024, 1024, 1024, 1024, 1024,
         16 * 1024, 1024 * 1024, 1024, 16 * 1024, G, 0);
    bn(stream, mh1, G, 16, 1024, 1, stats, part);

    gemm(stream, mh1, mp_w4, mp_b4, mfeat, 16, 128, 1024, 1024, 128, 128,
         16 * 1024, 1024 * 128, 128, 16 * 128, G, 0);
    bn(stream, mfeat, G, 16, 128, 1, stats, part);

    // ---- Generator axform -> x1g ; emit x_1
    axform(stream, mfeat, gw, xb, ah1, ah2, bigbuf, ybuf, stats, part, nullptr, x1g);
    gather_out_kernel<<<(unsigned)((T3 + 255) / 256), 256, 0, stream>>>(x1g, out, T3);

    // ---- Per-branch FPS of generated clouds into x_base[:, :512], then append x_partial
    fps_kernel<<<G * BSZ, 256, 0, stream>>>(x1g, 1024, 1024 * 3, NPS, xbase, 1024 * 3);
    {   long long t = (long long)G * BSZ * 512 * 3;
        fill_base_kernel<<<(unsigned)((t + 255) / 256), 256, 0, stream>>>(xpart, xbase, t); }

    // ---- Refinement axform (residual on x_base) -> x2g ; emit x_2
    axform(stream, mfeat, rw, xb, ah1, ah2, bigbuf, ybuf, stats, part, xbase, x2g);
    gather_out_kernel<<<(unsigned)((T3 + 255) / 256), 256, 0, stream>>>(x2g, out + T3, T3);
}
